// GatFCM_65592740544601
// MI455X (gfx1250) — compile-verified, matrix operand order checked by reading, not run on hardware
//
#include <hip/hip_runtime.h>
#include <hip/hip_bf16.h>

// GATv2Conv (heads=1) for MI455X / gfx1250.
// d_in order: 0=x[20000,1028] f32, 1=edge_index[2,320000] int,
//             2=Wl[1028,256] f32, 3=bl[256], 4=Wr[1028,256], 5=br[256],
//             6=att[256], 7=bias[256].  d_out: [20000,256] f32.

typedef __attribute__((ext_vector_type(16))) __bf16 v16bf;
typedef __attribute__((ext_vector_type(8)))  float  v8f;

#define N_NODES   20000
#define N_EDGES   320000
#define IN_DIM    1028
#define OUT_DIM   256
#define NEG_SLOPE 0.2f
#define KC        33                 // ceil(1028/32) K-chunks of 32 (zero padded)
#define NT        16                 // OUT_DIM/16 column tiles
#define MT        (N_NODES/16)       // 1250 row tiles (exact)
#define NE_TOT    (N_EDGES + N_NODES)

#if __has_builtin(__builtin_amdgcn_global_load_async_to_lds_b128)
#define HAVE_ASYNC_LDS 1
// Builtin signature (from hipcc diagnostic): arg0 = vector_size(16) int,
// __device__ (AS1) pointer, non-const; arg1 = LDS (AS3) pointer; then two imms.
typedef int v4i_raw __attribute__((vector_size(16)));
typedef v4i_raw __attribute__((address_space(1)))* gp128_t;
typedef v4i_raw __attribute__((address_space(3)))* lp128_t;
#else
#define HAVE_ASYNC_LDS 0
#endif

// ---------------------------------------------------------------------------
// Pack Wl/Wr (f32 [K,256]) into bf16 WMMA B-fragment layout.
// Entry id = ((w*16+nt)*KC + kc)*32 + lane ; each entry = 16 bf16 (one lane's
// B fragment for one 16x16x32 WMMA). 16-bit B 32x16 layout (mirror of A):
//   lane<16 : col n=lane,     elems 0..7 -> K 0..7,  8..15 -> K 16..23
//   lane>=16: col n=lane-16,  elems 0..7 -> K 8..15, 8..15 -> K 24..31
// ---------------------------------------------------------------------------
__global__ void pack_b_kernel(const float* __restrict__ Wl,
                              const float* __restrict__ Wr,
                              __bf16* __restrict__ Bpack) {
  int id = blockIdx.x * blockDim.x + threadIdx.x;
  if (id >= 2 * NT * KC * 32) return;
  const int lane  = id & 31;
  const int kc    = (id >> 5) % KC;
  const int nt    = (id / (32 * KC)) & (NT - 1);
  const int w     = id / (32 * KC * NT);
  const float* W  = w ? Wr : Wl;
  const int n     = nt * 16 + (lane & 15);
  const int khalf = lane >> 4;
  __bf16* dstp = Bpack + (size_t)id * 16;
#pragma unroll
  for (int e = 0; e < 16; ++e) {
    int kk = (e < 8) ? (khalf * 8 + e) : (16 + khalf * 8 + (e - 8));
    int k  = kc * 32 + kk;
    float v = (k < IN_DIM) ? W[(size_t)k * OUT_DIM + n] : 0.0f;
    dstp[e] = (__bf16)v;
  }
}

// ---------------------------------------------------------------------------
// Init: out = bias (broadcast), m = -inf, s = 0.
// ---------------------------------------------------------------------------
__global__ void init_kernel(float* __restrict__ out, const float* __restrict__ bias,
                            float* __restrict__ m, float* __restrict__ s) {
  int i = blockIdx.x * blockDim.x + threadIdx.x;
  if (i < N_NODES * OUT_DIM) out[i] = bias[i & (OUT_DIM - 1)];
  if (i < N_NODES) { m[i] = -3.4e38f; s[i] = 0.0f; }
}

// ---------------------------------------------------------------------------
// Fused GEMM: xl = x@Wl + bl, xr = x@Wr + br via v_wmma_f32_16x16x32_bf16.
// One block per 16-row tile:
//   1) async-copy the contiguous 16x1028 f32 slab of x into LDS
//   2) convert in-LDS to bf16 A-fragment layout
//   3) 8 waves x 2 job-pairs; each pair shares the A fragment and keeps two
//      independent accumulators (2-way WMMA ILP, half the ds_load traffic).
// ---------------------------------------------------------------------------
__global__ __launch_bounds__(256) void gemm_kernel(
    const float* __restrict__ x, const __bf16* __restrict__ Bpack,
    const float* __restrict__ bl, const float* __restrict__ br,
    float* __restrict__ xl, float* __restrict__ xr) {
  __shared__ __align__(16) float  Xraw[16 * IN_DIM];   // 65792 B (contiguous slab)
  __shared__ __align__(32) __bf16 Albs[KC * 32 * 16];  // 33792 B

  const int mt  = blockIdx.x;
  const int tid = threadIdx.x;
  const float* slab = x + (size_t)mt * 16 * IN_DIM;    // rows are contiguous

  // ---- Stage raw x slab into LDS (gfx1250 async copy if available) ----
#if HAVE_ASYNC_LDS
  for (int i = tid; i < (16 * IN_DIM) / 4; i += 256) {
    __builtin_amdgcn_global_load_async_to_lds_b128(
        (gp128_t)(void*)(slab + i * 4), (lp128_t)(void*)(&Xraw[i * 4]), 0, 0);
  }
  asm volatile("s_wait_asynccnt 0x0" ::: "memory");
#else
  for (int i = tid * 4; i < 16 * IN_DIM; i += 1024)
    *(float4*)&Xraw[i] = *(const float4*)&slab[i];
#endif
  __syncthreads();

  // ---- Convert to A-fragment layout: entry t=(kc,lane) -> 16 bf16 ----
  for (int t = tid; t < KC * 32; t += 256) {
    const int kc    = t >> 5;
    const int lane  = t & 31;
    const int khalf = lane >> 4;
    const float* xrow = &Xraw[(lane & 15) * IN_DIM];
    __bf16* dstp = &Albs[t * 16];
#pragma unroll
    for (int e = 0; e < 16; ++e) {
      int kk = (e < 8) ? (khalf * 8 + e) : (16 + khalf * 8 + (e - 8));
      int k  = kc * 32 + kk;
      dstp[e] = (__bf16)((k < IN_DIM) ? xrow[k] : 0.0f);
    }
  }
  __syncthreads();

  const int wave = tid >> 5;
  const int lane = tid & 31;
  const __bf16* afrag = &Albs[lane * 16];

  for (int jp = 0; jp < 2; ++jp) {                 // job-pair loop
    const int j0 = wave * 4 + jp * 2;              // job = w*16 + nt
    v8f c0 = {}, c1 = {};
    const __bf16* bp0 = Bpack + (size_t)j0 * KC * 512 + lane * 16;
    const __bf16* bp1 = bp0 + (size_t)KC * 512;
    for (int kc = 0; kc < KC; ++kc) {
      v16bf a  = *(const v16bf*)(afrag + kc * 512);
      v16bf b0 = *(const v16bf*)(bp0 + (size_t)kc * 512);
      v16bf b1 = *(const v16bf*)(bp1 + (size_t)kc * 512);
      c0 = __builtin_amdgcn_wmma_f32_16x16x32_bf16(false, a, false, b0,
                                                   (short)0, c0, false, false);
      c1 = __builtin_amdgcn_wmma_f32_16x16x32_bf16(false, a, false, b1,
                                                   (short)0, c1, false, false);
    }
    // Epilogue: row = mt*16 + r + 8*(lane>>4), col = nt*16 + (lane&15)
    const int rbase = mt * 16 + ((lane >> 4) << 3);
#pragma unroll
    for (int q = 0; q < 2; ++q) {
      const int j    = j0 + q;
      const int wsel = j >> 4;
      const int col  = (j & 15) * 16 + (lane & 15);
      const float biasv = (wsel == 0 ? bl : br)[col];
      float*      outp  = (wsel == 0 ? xl : xr);
      const v8f&  c     = q ? c1 : c0;
#pragma unroll
      for (int r = 0; r < 8; ++r)
        outp[(size_t)(rbase + r) * OUT_DIM + col] = c[r] + biasv;
    }
  }
}

// ---------------------------------------------------------------------------
// Edge logits: e = att . leakyrelu(xl[src] + xr[dst]); atomic f32 max into m.
// One wave32 per edge; lane covers 8 of 256 dims.
// ---------------------------------------------------------------------------
__global__ __launch_bounds__(256) void edge_logit_kernel(
    const int* __restrict__ ei, const float* __restrict__ xl,
    const float* __restrict__ xr, const float* __restrict__ att,
    float* __restrict__ ebuf, float* __restrict__ m) {
  const int eid = blockIdx.x * 8 + (threadIdx.x >> 5);
  if (eid >= NE_TOT) return;
  const int lane = threadIdx.x & 31;
  int src, dst;
  if (eid < N_EDGES) { src = ei[eid]; dst = ei[N_EDGES + eid]; }
  else               { src = dst = eid - N_EDGES; }

  const float4* pl = (const float4*)(xl + (size_t)src * OUT_DIM) + lane * 2;
  const float4* pr = (const float4*)(xr + (size_t)dst * OUT_DIM) + lane * 2;
  const float4* pa = (const float4*)att + lane * 2;
  float sum = 0.0f;
#pragma unroll
  for (int q = 0; q < 2; ++q) {
    float4 l4 = pl[q], r4 = pr[q], a4 = pa[q];
    float v;
    v = l4.x + r4.x; sum += a4.x * (v > 0.f ? v : NEG_SLOPE * v);
    v = l4.y + r4.y; sum += a4.y * (v > 0.f ? v : NEG_SLOPE * v);
    v = l4.z + r4.z; sum += a4.z * (v > 0.f ? v : NEG_SLOPE * v);
    v = l4.w + r4.w; sum += a4.w * (v > 0.f ? v : NEG_SLOPE * v);
  }
#pragma unroll
  for (int off = 16; off > 0; off >>= 1) sum += __shfl_down(sum, off, 32);
  if (lane == 0) {
    ebuf[eid] = sum;
    __hip_atomic_fetch_max(&m[dst], sum, __ATOMIC_RELAXED,
                           __HIP_MEMORY_SCOPE_AGENT);
  }
}

// ---------------------------------------------------------------------------
// a = exp(e - m[dst]); s[dst] += a.
// ---------------------------------------------------------------------------
__global__ void edge_exp_kernel(const int* __restrict__ ei,
                                const float* __restrict__ ebuf,
                                const float* __restrict__ m,
                                float* __restrict__ abuf,
                                float* __restrict__ s) {
  int eid = blockIdx.x * blockDim.x + threadIdx.x;
  if (eid >= NE_TOT) return;
  int dst = (eid < N_EDGES) ? ei[N_EDGES + eid] : eid - N_EDGES;
  float a = __expf(ebuf[eid] - m[dst]);
  abuf[eid] = a;
  atomicAdd(&s[dst], a);
}

// ---------------------------------------------------------------------------
// out[dst] += (a/s[dst]) * xl[src], wave32 per edge, 8 f32 atomics per lane.
// ---------------------------------------------------------------------------
__global__ __launch_bounds__(256) void scatter_kernel(
    const int* __restrict__ ei, const float* __restrict__ xl,
    const float* __restrict__ abuf, const float* __restrict__ s,
    float* __restrict__ out) {
  const int eid = blockIdx.x * 8 + (threadIdx.x >> 5);
  if (eid >= NE_TOT) return;
  const int lane = threadIdx.x & 31;
  int src, dst;
  if (eid < N_EDGES) { src = ei[eid]; dst = ei[N_EDGES + eid]; }
  else               { src = dst = eid - N_EDGES; }
  const float alpha = abuf[eid] / s[dst];
  const float* pl = xl + (size_t)src * OUT_DIM + lane * 8;
  float*       po = out + (size_t)dst * OUT_DIM + lane * 8;
#pragma unroll
  for (int q = 0; q < 8; ++q) atomicAdd(&po[q], alpha * pl[q]);
}

// ---------------------------------------------------------------------------
extern "C" void kernel_launch(void* const* d_in, const int* in_sizes, int n_in,
                              void* d_out, int out_size, void* d_ws, size_t ws_size,
                              hipStream_t stream) {
  const float* x    = (const float*)d_in[0];
  const int*   ei   = (const int*)d_in[1];   // [2, N_EDGES], row0=src row1=dst
  const float* Wl   = (const float*)d_in[2];
  const float* bl   = (const float*)d_in[3];
  const float* Wr   = (const float*)d_in[4];
  const float* br   = (const float*)d_in[5];
  const float* att  = (const float*)d_in[6];
  const float* bias = (const float*)d_in[7];
  float* out = (float*)d_out;

  // Workspace carve-up (floats).
  float* xl   = (float*)d_ws;
  float* xr   = xl + (size_t)N_NODES * OUT_DIM;
  float* ebuf = xr + (size_t)N_NODES * OUT_DIM;
  float* abuf = ebuf + NE_TOT;
  float* mbuf = abuf + NE_TOT;
  float* sbuf = mbuf + N_NODES;
  __bf16* Bpack = (__bf16*)(sbuf + N_NODES);   // 2*16*33*32*16 bf16 ~ 1.06 MB

  const int packN = 2 * NT * KC * 32;
  pack_b_kernel<<<(packN + 255) / 256, 256, 0, stream>>>(Wl, Wr, Bpack);
  init_kernel<<<(N_NODES * OUT_DIM + 255) / 256, 256, 0, stream>>>(out, bias, mbuf, sbuf);
  gemm_kernel<<<MT, 256, 0, stream>>>(x, Bpack, bl, br, xl, xr);

  const int eblocks = (NE_TOT + 7) / 8;
  edge_logit_kernel<<<eblocks, 256, 0, stream>>>(ei, xl, xr, att, ebuf, mbuf);
  edge_exp_kernel<<<(NE_TOT + 255) / 256, 256, 0, stream>>>(ei, ebuf, mbuf, abuf, sbuf);
  scatter_kernel<<<eblocks, 256, 0, stream>>>(ei, xl, abuf, sbuf, out);
}
